// PatternMatching_3599182594547
// MI455X (gfx1250) — matrix operand order repaired and checked by reading, not compiled, not run on hardware
//
#include <hip/hip_runtime.h>
#include <hip/hip_fp16.h>
#include <math.h>

#define Bsz   64
#define SZ    256
#define S1    257
#define LEN   32
#define NEGF  (-1.0e9f)
#define KA    256   // alpha GEMM K  (source states i = 0..255; row 256 handled separately)
#define NAP   272   // alpha GEMM N  (targets j = 0..256, padded to 17 tiles)
#define KBP   288   // beta  GEMM K  (targets j = 0..256, padded to 9 k-chunks)
#define NBP   256   // beta  GEMM N  (sources i = 0..255)

#define SLAB_HALFS (NAP * 32)             // max k-chunk slab: 272 cols x 32 f16 = 17408 B
#define DSM_BYTES  (2 * SLAB_HALFS * 2 + Bsz * KBP * 2)  // 2 slabs + A staging = 71680 B

typedef __attribute__((ext_vector_type(16))) _Float16 v16h;
typedef __attribute__((ext_vector_type(8)))  _Float16 v8h;
typedef __attribute__((ext_vector_type(8)))  float    v8f;
typedef __attribute__((ext_vector_type(4)))  unsigned int u32x4;
typedef __attribute__((ext_vector_type(8)))  int i32x8;
typedef __attribute__((ext_vector_type(4)))  int i32x4;

__device__ __forceinline__ float lse2f(float x, float y) {
  float m = fmaxf(x, y);
  if (m <= -0.5e9f) return m;  // both effectively -inf
  return m + __logf(__expf(x - m) + __expf(y - m));
}

// ---- TDM: async DMA of one k-chunk slab (ncols rows of 32 contiguous f16) ----
// 2D descriptor: X = 32 contiguous elements, Y = ncols lines, line stride = Kelems.
__device__ __forceinline__ void tdm_issue_slab(const _Float16* gbase, int kc,
                                               const _Float16* ldsbuf,
                                               int Kelems, int ncols) {
  unsigned long long ga =
      (unsigned long long)(uintptr_t)gbase + (unsigned long long)kc * 32u * 2u;
  unsigned ldsoff = (unsigned)(uintptr_t)ldsbuf;  // low 32 bits = LDS byte address
  u32x4 g0;
  g0[0] = 1u;                                         // count=1, user descriptor
  g0[1] = ldsoff;                                     // lds_addr
  g0[2] = (unsigned)(ga & 0xFFFFFFFFull);             // global_addr[31:0]
  g0[3] = (unsigned)((ga >> 32) & 0x1FFFFFFull)       // global_addr[56:32]
          | 0x80000000u;                              // type = 2 ("image")
  i32x8 g1;
  g1[0] = (1 << 16);                                  // data_size = 1 -> 2 bytes
  g1[1] = (int)((32u & 0xFFFFu) << 16);               // tensor_dim0 = 32 (lo16)
  g1[2] = (int)(((unsigned)ncols & 0xFFFFu) << 16);   // dim0 hi=0 | tensor_dim1 lo
  g1[3] = (int)(32u << 16);                           // dim1 hi=0 | tile_dim0 = 32
  g1[4] = ncols;                                      // tile_dim1 = ncols, tile_dim2 = 0
  g1[5] = Kelems;                                     // tensor_dim0_stride (lo32)
  g1[6] = 0;                                          // stride0 hi | stride1 lo (unused, 2D)
  g1[7] = 0;
  i32x4 z4;
  z4[0] = z4[1] = z4[2] = z4[3] = 0;
#if defined(__clang_major__) && (__clang_major__ >= 23)
  i32x8 z8;
  for (int e = 0; e < 8; ++e) z8[e] = 0;
  __builtin_amdgcn_tensor_load_to_lds(g0, g1, z4, z4, z8, 0);
#else
  __builtin_amdgcn_tensor_load_to_lds(g0, g1, z4, z4, 0);
#endif
}

// -------- cs[b,ch,k] = prefix sums of log_softmax(logits, axis=ch) ----------
__global__ void k_prep_cs(const float* __restrict__ logits, float* __restrict__ cs) {
  int b = blockIdx.x, k = threadIdx.x;  // 256 threads
  __shared__ float sc[SZ];
  float l0 = logits[(b * 3 + 0) * SZ + k];
  float l1 = logits[(b * 3 + 1) * SZ + k];
  float l2 = logits[(b * 3 + 2) * SZ + k];
  float m = fmaxf(l0, fmaxf(l1, l2));
  float lse = m + __logf(__expf(l0 - m) + __expf(l1 - m) + __expf(l2 - m));
  float C[3] = {l0 - lse, l1 - lse, l2 - lse};
  for (int ch = 0; ch < 3; ++ch) {
    sc[k] = C[ch];
    __syncthreads();
    for (int off = 1; off < SZ; off <<= 1) {
      float v = sc[k];
      if (k >= off) v += sc[k - off];
      __syncthreads();
      sc[k] = v;
      __syncthreads();
    }
    if (k == 0) cs[(b * 3 + ch) * S1 + 0] = 0.f;
    cs[(b * 3 + ch) * S1 + k + 1] = sc[k];
    __syncthreads();
  }
}

// -------- WaT[c][j][i] = exp(Q[c,i,j]-colmax_j) masked i<j (K-contiguous) ----
__global__ void k_prep_wa(const float* __restrict__ Q, float* __restrict__ colA,
                          __half* WaTh) {
  _Float16* WaT = (_Float16*)WaTh;
  int n = blockIdx.x, ch = blockIdx.y, k = threadIdx.x;  // n=j (0..271), k=i
  __shared__ float sb[KA];
  __shared__ float cmS;
  bool valid = (n <= SZ) && (k < n);
  float q = valid ? Q[((size_t)ch * S1 + k) * S1 + n] : -3.0e38f;
  sb[k] = q;
  __syncthreads();
  if (k == 0) {
    float mx = -3.0e38f;
    for (int t = 0; t < KA; ++t) mx = fmaxf(mx, sb[t]);
    cmS = (mx > -1.0e38f) ? mx : 0.f;
    colA[ch * NAP + n] = cmS;
  }
  __syncthreads();
  WaT[((size_t)(ch * NAP + n)) * KA + k] =
      valid ? (_Float16)__expf(q - cmS) : (_Float16)0.f;
}

// -------- WbT[c][i][j] = exp(Q[c,i,j]-rowmax_i) masked j>i (K-contiguous) ----
__global__ void k_prep_wb(const float* __restrict__ Q, float* __restrict__ rowB,
                          __half* WbTh) {
  _Float16* WbT = (_Float16*)WbTh;
  int n = blockIdx.x, ch = blockIdx.y, k = threadIdx.x;  // n=i (0..255), k=j (0..287)
  __shared__ float sb[KBP];
  __shared__ float rmS;
  bool valid = (k <= SZ) && (k > n);
  float q = valid ? Q[((size_t)ch * S1 + n) * S1 + k] : -3.0e38f;
  sb[k] = q;
  __syncthreads();
  if (k == 0) {
    float mx = -3.0e38f;
    for (int t = 0; t < KBP; ++t) mx = fmaxf(mx, sb[t]);
    rmS = (mx > -1.0e38f) ? mx : 0.f;
    rowB[ch * NBP + n] = rmS;
  }
  __syncthreads();
  WbT[((size_t)(ch * NBP + n)) * KBP + k] =
      valid ? (_Float16)__expf(q - rmS) : (_Float16)0.f;
}

// -------- initial alpha/beta state ------------------------------------------
__global__ void k_init(const int* __restrict__ ls, float* la, float* lb,
                       float* na, float* nb, float* aout, float* bout) {
  int b = blockIdx.x, j = threadIdx.x;  // 288 threads
  if (j < S1) {
    float av = (j == 0) ? 0.f : NEGF;
    float bv = (j == ls[b]) ? 0.f : NEGF;
    la[b * S1 + j] = av;
    lb[b * S1 + j] = bv;
    aout[((size_t)b * (LEN + 1) + 0) * S1 + j] = av;
    bout[((size_t)b * (LEN + 1) + LEN) * S1 + j] = bv;
  }
  if (j == 0) { na[b] = 0.f; nb[b] = 0.f; }
}

// -------- one DP step: block 0 = alpha, block 1 = beta ----------------------
// W k-chunk slabs are staged global->LDS by the Tensor Data Mover, double
// buffered (issue-ahead 2, s_wait_tensorcnt + workgroup barrier to publish).
__global__ void __launch_bounds__(1024) k_step(
    const int* __restrict__ pattern, int step,
    const float* __restrict__ Q, const float* __restrict__ cs,
    const float* __restrict__ colA, const float* __restrict__ rowB,
    const __half* WaTh, const __half* WbTh,
    float* la, float* lb, float* na, float* nb, float* aout, float* bout) {
  extern __shared__ char dsm[];
  _Float16* wS0 = (_Float16*)dsm;                 // slab buffer 0 (17408 B)
  _Float16* wS1 = wS0 + SLAB_HALFS;               // slab buffer 1 (17408 B)
  _Float16* aS  = wS1 + SLAB_HALFS;               // A staging (36864 B)
  __shared__ float red[Bsz][16];
  __shared__ float shiftS[Bsz], auxS[Bsz], gmaxS[Bsz], lseS[Bsz], noldS[Bsz];

  int tid = threadIdx.x;
  int bb = tid >> 4, slot = tid & 15;
  int lane = tid & 31, wave = tid >> 5;

  if (blockIdx.x == 0) {
    // ================================ ALPHA ================================
    int c = pattern[step];
    const _Float16* WaT = (const _Float16*)WaTh + (size_t)c * NAP * KA;
    const int NTn = NAP / 16;  // 17
    const int NK = KA / 32;    // 8
    // kick off TDM for slabs 0 and 1 (overlaps with the exp staging below)
    if (wave == 0) {
      tdm_issue_slab(WaT, 0, wS0, KA, NAP);
      tdm_issue_slab(WaT, 1, wS1, KA, NAP);
    }
    // phase A: per-batch shift + f16 staging of v = exp(la - cs - shift)
    float mx = -3.0e38f;
    for (int i = slot; i < SZ; i += 16)
      mx = fmaxf(mx, la[bb * S1 + i] - cs[(bb * 3 + c) * S1 + i]);
    red[bb][slot] = mx;
    __syncthreads();
    if (slot == 0) {
      float m2 = red[bb][0];
      for (int t = 1; t < 16; ++t) m2 = fmaxf(m2, red[bb][t]);
      shiftS[bb] = m2;
      auxS[bb]  = la[bb * S1 + SZ];  // la[b][256] for the special row
      noldS[bb] = na[bb];
    }
    __syncthreads();
    for (int idx = tid; idx < Bsz * KA; idx += 1024) {
      int m = idx >> 8, i = idx & 255;
      aS[m * KA + i] =
          (_Float16)__expf(la[m * S1 + i] - cs[(m * 3 + c) * S1 + i] - shiftS[m]);
    }
    __syncthreads();
    // phase B: slab-pipelined GEMM (64x256)x(256x272) via WMMA
    v8f acc[3];
#pragma unroll
    for (int r = 0; r < 3; ++r)
      for (int e = 0; e < 8; ++e) acc[r][e] = 0.f;
    for (int kc = 0; kc < NK; ++kc) {
      if (wave == 0) {
        if (kc < NK - 1) __builtin_amdgcn_s_wait_tensorcnt(1);
        else             __builtin_amdgcn_s_wait_tensorcnt(0);
      }
      __syncthreads();  // slab kc visible to all waves
      const _Float16* wbuf = (kc & 1) ? wS1 : wS0;
#pragma unroll
      for (int r = 0; r < 3; ++r) {
        int t = wave + 32 * r;
        if (t < 4 * NTn) {
          int mt = t / NTn, nt = t % NTn;
          int row = mt * 16 + (lane & 15);
          int koff = (lane < 16) ? 0 : 16;
          const _Float16* ap = aS + row * KA + kc * 32 + koff;
          v8h alo = *(const v8h*)(ap);
          v8h ahi = *(const v8h*)(ap + 8);
          v16h af = __builtin_shufflevector(alo, ahi, 0, 1, 2, 3, 4, 5, 6, 7, 8,
                                            9, 10, 11, 12, 13, 14, 15);
          int n = nt * 16 + (lane & 15);
          const _Float16* bp = wbuf + n * 32 + ((lane < 16) ? 0 : 16);
          v8h b0 = *(const v8h*)(bp);
          v8h b1 = *(const v8h*)(bp + 8);
          v16h bf = __builtin_shufflevector(b0, b1, 0, 1, 2, 3, 4, 5, 6, 7, 8, 9,
                                            10, 11, 12, 13, 14, 15);
          acc[r] = __builtin_amdgcn_wmma_f32_16x16x32_f16(
              false, af, false, bf, (short)0, acc[r], false, false);
        }
      }
      __syncthreads();  // all waves done with buffer (kc%2) before re-fill
      if (wave == 0 && kc + 2 < NK)
        tdm_issue_slab(WaT, kc + 2, (kc & 1) ? wS1 : wS0, KA, NAP);
    }
    // combine tiles -> raw log-alpha
#pragma unroll
    for (int r = 0; r < 3; ++r) {
      int t = wave + 32 * r;
      if (t < 4 * NTn) {
        int mt = t / NTn, nt = t % NTn;
        int ncol = nt * 16 + (lane & 15);
        int mrow0 = mt * 16 + ((lane >> 4) << 3);
        if (ncol <= SZ) {
          for (int g = 0; g < 8; ++g) {
            int m = mrow0 + g;
            float s = acc[r][g];
            float gl = (s > 0.f)
                           ? (cs[(m * 3 + c) * S1 + ncol] + shiftS[m] +
                              colA[c * NAP + ncol] + __logf(s))
                           : NEGF;
            float szt = auxS[m] + Q[((size_t)c * S1 + SZ) * S1 + ncol];
            aout[((size_t)m * (LEN + 1) + step + 1) * S1 + ncol] = lse2f(gl, szt);
          }
        }
      }
    }
    __syncthreads();
    // phase C: online renormalization (logsumexp over j), emit a+n
    float m1 = -3.0e38f;
    for (int j = slot; j < S1; j += 16)
      m1 = fmaxf(m1, aout[((size_t)bb * (LEN + 1) + step + 1) * S1 + j]);
    red[bb][slot] = m1;
    __syncthreads();
    if (slot == 0) {
      float m2 = red[bb][0];
      for (int t = 1; t < 16; ++t) m2 = fmaxf(m2, red[bb][t]);
      gmaxS[bb] = m2;
    }
    __syncthreads();
    float sm = 0.f;
    for (int j = slot; j < S1; j += 16)
      sm += __expf(aout[((size_t)bb * (LEN + 1) + step + 1) * S1 + j] - gmaxS[bb]);
    red[bb][slot] = sm;
    __syncthreads();
    if (slot == 0) {
      float s2 = 0.f;
      for (int t = 0; t < 16; ++t) s2 += red[bb][t];
      lseS[bb] = gmaxS[bb] + __logf(s2);
    }
    __syncthreads();
    for (int j = slot; j < S1; j += 16) {
      size_t o = ((size_t)bb * (LEN + 1) + step + 1) * S1 + j;
      float raw = aout[o];
      la[bb * S1 + j] = raw - lseS[bb];
      aout[o] = raw + noldS[bb];
    }
    if (slot == 0) na[bb] = noldS[bb] + lseS[bb];
  } else {
    // ================================ BETA =================================
    int c = pattern[LEN - 1 - step];
    int pos = LEN - 1 - step;
    const _Float16* WbT = (const _Float16*)WbTh + (size_t)c * NBP * KBP;
    const int NTn = NBP / 16;  // 16
    const int NK = KBP / 32;   // 9
    if (wave == 0) {
      tdm_issue_slab(WbT, 0, wS0, KBP, NBP);
      tdm_issue_slab(WbT, 1, wS1, KBP, NBP);
    }
    float mx = -3.0e38f;
    for (int j = slot; j < S1; j += 16)
      mx = fmaxf(mx, lb[bb * S1 + j] + cs[(bb * 3 + c) * S1 + j]);
    red[bb][slot] = mx;
    __syncthreads();
    if (slot == 0) {
      float m2 = red[bb][0];
      for (int t = 1; t < 16; ++t) m2 = fmaxf(m2, red[bb][t]);
      shiftS[bb] = m2;
      noldS[bb] = nb[bb];
    }
    __syncthreads();
    for (int idx = tid; idx < Bsz * KBP; idx += 1024) {
      int m = idx / KBP, j = idx % KBP;
      aS[m * KBP + j] =
          (j < S1) ? (_Float16)__expf(lb[m * S1 + j] + cs[(m * 3 + c) * S1 + j] -
                                      shiftS[m])
                   : (_Float16)0.f;
    }
    __syncthreads();
    v8f acc[3];
#pragma unroll
    for (int r = 0; r < 3; ++r)
      for (int e = 0; e < 8; ++e) acc[r][e] = 0.f;
    for (int kc = 0; kc < NK; ++kc) {
      if (wave == 0) {
        if (kc < NK - 1) __builtin_amdgcn_s_wait_tensorcnt(1);
        else             __builtin_amdgcn_s_wait_tensorcnt(0);
      }
      __syncthreads();
      const _Float16* wbuf = (kc & 1) ? wS1 : wS0;
#pragma unroll
      for (int r = 0; r < 3; ++r) {
        int t = wave + 32 * r;
        if (t < 4 * NTn) {
          int mt = t / NTn, nt = t % NTn;
          int row = mt * 16 + (lane & 15);
          int koff = (lane < 16) ? 0 : 16;
          const _Float16* ap = aS + row * KBP + kc * 32 + koff;
          v8h alo = *(const v8h*)(ap);
          v8h ahi = *(const v8h*)(ap + 8);
          v16h af = __builtin_shufflevector(alo, ahi, 0, 1, 2, 3, 4, 5, 6, 7, 8,
                                            9, 10, 11, 12, 13, 14, 15);
          int n = nt * 16 + (lane & 15);
          const _Float16* bp = wbuf + n * 32 + ((lane < 16) ? 0 : 16);
          v8h b0 = *(const v8h*)(bp);
          v8h b1 = *(const v8h*)(bp + 8);
          v16h bf = __builtin_shufflevector(b0, b1, 0, 1, 2, 3, 4, 5, 6, 7, 8, 9,
                                            10, 11, 12, 13, 14, 15);
          acc[r] = __builtin_amdgcn_wmma_f32_16x16x32_f16(
              false, af, false, bf, (short)0, acc[r], false, false);
        }
      }
      __syncthreads();
      if (wave == 0 && kc + 2 < NK)
        tdm_issue_slab(WbT, kc + 2, (kc & 1) ? wS1 : wS0, KBP, NBP);
    }
#pragma unroll
    for (int r = 0; r < 3; ++r) {
      int t = wave + 32 * r;
      if (t < 4 * NTn) {
        int mt = t / NTn, nt = t % NTn;
        int ncol = nt * 16 + (lane & 15);  // i
        int mrow0 = mt * 16 + ((lane >> 4) << 3);
        for (int g = 0; g < 8; ++g) {
          int m = mrow0 + g;
          float s = acc[r][g];
          float rr = (s > 0.f)
                         ? (-cs[(m * 3 + c) * S1 + ncol] + rowB[c * NBP + ncol] +
                            shiftS[m] + __logf(s))
                         : NEGF;
          bout[((size_t)m * (LEN + 1) + pos) * S1 + ncol] = rr;
        }
      }
    }
    __syncthreads();
    // special row i = 256: lse_j(Q[c,256,j] + lb[b,j])
    float m256 = -3.0e38f;
    for (int j = slot; j < S1; j += 16)
      m256 = fmaxf(m256, Q[((size_t)c * S1 + SZ) * S1 + j] + lb[bb * S1 + j]);
    red[bb][slot] = m256;
    __syncthreads();
    if (slot == 0) {
      float m2 = red[bb][0];
      for (int t = 1; t < 16; ++t) m2 = fmaxf(m2, red[bb][t]);
      gmaxS[bb] = m2;
    }
    __syncthreads();
    float s256 = 0.f;
    for (int j = slot; j < S1; j += 16)
      s256 += __expf(Q[((size_t)c * S1 + SZ) * S1 + j] + lb[bb * S1 + j] -
                     gmaxS[bb]);
    red[bb][slot] = s256;
    __syncthreads();
    if (slot == 0) {
      float s2 = 0.f;
      for (int t = 0; t < 16; ++t) s2 += red[bb][t];
      bout[((size_t)bb * (LEN + 1) + pos) * S1 + SZ] = gmaxS[bb] + __logf(s2);
    }
    __syncthreads();
    // renormalize over i = 0..256
    float m1 = -3.0e38f;
    for (int i = slot; i < S1; i += 16)
      m1 = fmaxf(m1, bout[((size_t)bb * (LEN + 1) + pos) * S1 + i]);
    red[bb][slot] = m1;
    __syncthreads();
    if (slot == 0) {
      float m2 = red[bb][0];
      for (int t = 1; t < 16; ++t) m2 = fmaxf(m2, red[bb][t]);
      gmaxS[bb] = m2;
    }
    __syncthreads();
    float sm = 0.f;
    for (int i = slot; i < S1; i += 16)
      sm += __expf(bout[((size_t)bb * (LEN + 1) + pos) * S1 + i] - gmaxS[bb]);
    red[bb][slot] = sm;
    __syncthreads();
    if (slot == 0) {
      float s2 = 0.f;
      for (int t = 0; t < 16; ++t) s2 += red[bb][t];
      lseS[bb] = gmaxS[bb] + __logf(s2);
    }
    __syncthreads();
    for (int i = slot; i < S1; i += 16) {
      size_t o = ((size_t)bb * (LEN + 1) + pos) * S1 + i;
      float raw = bout[o];
      lb[bb * S1 + i] = raw - lseS[bb];
      bout[o] = raw + noldS[bb];
    }
    if (slot == 0) nb[bb] = noldS[bb] + lseS[bb];
  }
}

// -------- M[b] = a[b, LEN, ls[b]]; also write ls as float -------------------
__global__ void k_M(const int* __restrict__ ls, const float* __restrict__ aout,
                    float* Mb, float* outM, float* outls) {
  int b = threadIdx.x;
  float m = aout[((size_t)b * (LEN + 1) + LEN) * S1 + ls[b]];
  Mb[b] = m;
  outM[b] = m;
  outls[b] = (float)ls[b];
}

// -------- ll = a + b ; p = a + b - M ----------------------------------------
__global__ void k_fin(const float* __restrict__ aout, const float* __restrict__ bout,
                      const float* __restrict__ Mb, float* ll, float* p) {
  size_t idx = (size_t)blockIdx.x * blockDim.x + threadIdx.x;
  if (idx >= (size_t)Bsz * (LEN + 1) * S1) return;
  int b = (int)(idx / ((LEN + 1) * S1));
  float v = aout[idx] + bout[idx];
  ll[idx] = v;
  p[idx] = v - Mb[b];
}

// -------- t[b,l,i,j] = a[b,l,i] + b[b,l+1,j] + P + Q - M  (541 MB stream) ---
__global__ void k_t(const int* __restrict__ pattern, const float* __restrict__ Q,
                    const float* __restrict__ cs, const float* __restrict__ aout,
                    const float* __restrict__ bout, const float* __restrict__ Mb,
                    float* __restrict__ t) {
  int rowid = blockIdx.x;
  int i = rowid % S1;
  int l = (rowid / S1) % LEN;
  int b = rowid / (S1 * LEN);
  int c = pattern[l];
  int j = threadIdx.x;
  __shared__ float aiS, MiS, ciS;
  if (j == 0) {
    aiS = aout[((size_t)b * (LEN + 1) + l) * S1 + i];
    MiS = Mb[b];
    ciS = cs[(b * 3 + c) * S1 + i];
  }
  __syncthreads();
  if (j < S1) {
    float bv = bout[((size_t)b * (LEN + 1) + l + 1) * S1 + j];
    float q = Q[((size_t)c * S1 + i) * S1 + j];
    float Pv = (j > i) ? (cs[(b * 3 + c) * S1 + j] - ciS)
                       : ((i == SZ) ? 0.f : NEGF);
    float val = aiS + bv + Pv + q - MiS;
    __builtin_nontemporal_store(val, &t[((size_t)rowid) * S1 + j]);
  }
}

extern "C" void kernel_launch(void* const* d_in, const int* in_sizes, int n_in,
                              void* d_out, int out_size, void* d_ws, size_t ws_size,
                              hipStream_t stream) {
  (void)in_sizes; (void)n_in; (void)out_size; (void)ws_size;
  const float* logits = (const float*)d_in[0];
  const float* Q = (const float*)d_in[1];
  const int* pattern = (const int*)d_in[2];
  const int* ls = (const int*)d_in[3];

  float* out = (float*)d_out;
  const size_t nABl = (size_t)Bsz * (LEN + 1) * S1;  // 542,784
  float* outM = out;
  float* outll = out + 64;
  float* aout = outll + nABl;   // 'a' output doubles as live alpha buffer
  float* bout = aout + nABl;    // 'b' output doubles as live beta buffer
  float* outls = bout + nABl;
  float* outp = outls + 64;
  float* outt = outp + nABl;

  char* w = (char*)d_ws;
  auto alignup = [](size_t x) { return (x + 255) & ~(size_t)255; };
  size_t off = 0;
  float* cs = (float*)(w + off);   off = alignup(off + sizeof(float) * Bsz * 3 * S1);
  float* colA = (float*)(w + off); off = alignup(off + sizeof(float) * 3 * NAP);
  float* rowB = (float*)(w + off); off = alignup(off + sizeof(float) * 3 * NBP);
  __half* WaT = (__half*)(w + off); off = alignup(off + sizeof(__half) * 3 * NAP * KA);
  __half* WbT = (__half*)(w + off); off = alignup(off + sizeof(__half) * 3 * NBP * KBP);
  float* la = (float*)(w + off);   off = alignup(off + sizeof(float) * Bsz * S1);
  float* lb = (float*)(w + off);   off = alignup(off + sizeof(float) * Bsz * S1);
  float* na = (float*)(w + off);   off = alignup(off + sizeof(float) * Bsz);
  float* nb = (float*)(w + off);   off = alignup(off + sizeof(float) * Bsz);
  float* Mb = (float*)(w + off);   off = alignup(off + sizeof(float) * Bsz);

  // allow > default dynamic LDS for the TDM-staged step kernel
  (void)hipFuncSetAttribute((const void*)k_step,
                            hipFuncAttributeMaxDynamicSharedMemorySize, DSM_BYTES);

  k_prep_cs<<<Bsz, SZ, 0, stream>>>(logits, cs);
  k_prep_wa<<<dim3(NAP, 3), KA, 0, stream>>>(Q, colA, WaT);
  k_prep_wb<<<dim3(NBP, 3), KBP, 0, stream>>>(Q, rowB, WbT);
  k_init<<<Bsz, KBP, 0, stream>>>(ls, la, lb, na, nb, aout, bout);
  for (int s = 0; s < LEN; ++s) {
    k_step<<<2, 1024, DSM_BYTES, stream>>>(pattern, s, Q, cs, colA, rowB, WaT,
                                           WbT, la, lb, na, nb, aout, bout);
  }
  k_M<<<1, Bsz, 0, stream>>>(ls, aout, Mb, outM, outls);
  int nfin = (int)((nABl + 255) / 256);
  k_fin<<<nfin, 256, 0, stream>>>(aout, bout, Mb, outll, outp);
  k_t<<<Bsz * LEN * S1, KBP, 0, stream>>>(pattern, Q, cs, aout, bout, Mb, outt);
}
